// RTDETRTransformer_26757646254624
// MI455X (gfx1250) — matrix operand order repaired and checked by reading, not compiled, not run on hardware
//
#include <hip/hip_runtime.h>

// ---------------------------------------------------------------------------
// RT-DETR decoder (6 layers) for gfx1250.
// Dense GEMMs: LDS-staged 64x64 block tiles, v_wmma_f32_16x16x32_f16 with
// f32 accumulate, weights pre-transposed to N x K f16. B tiles arrive via the
// Tensor Data Mover (tensor_load_to_lds + s_wait_tensorcnt) when the builtin
// is available, else via cooperative vector loads.
// ---------------------------------------------------------------------------

typedef __attribute__((ext_vector_type(16))) _Float16 v16h;
typedef __attribute__((ext_vector_type(8)))  _Float16 v8h;
typedef __attribute__((ext_vector_type(8)))  float    v8f;
typedef __attribute__((ext_vector_type(4)))  float    v4f;
typedef __attribute__((ext_vector_type(4)))  unsigned int v4u;
typedef __attribute__((ext_vector_type(8)))  int      v8i;
typedef __attribute__((ext_vector_type(4)))  int      v4i;

#define NLAYER 6
#define BS     16
#define NQ     300
#define DMODEL 256
#define NHEAD  8
#define HD     32
#define NLVL   3
#define NPT    4
#define DFF    1024
#define NCLS   80
#define LVTOT  8400
#define MROWS  (BS * NQ)          // 4800

#if defined(__HIP_DEVICE_COMPILE__) && __has_builtin(__builtin_amdgcn_tensor_load_to_lds) && __has_builtin(__builtin_amdgcn_s_wait_tensorcnt)
#define HAVE_TDM 1
#else
#define HAVE_TDM 0
#endif

// ---------------------------------------------------------------------------
// weight conversion: fp32 [mat][K][N] -> f16 transposed [mat][N][K]
// ---------------------------------------------------------------------------
__global__ void cvt_t_kernel(const float* __restrict__ src,
                             _Float16* __restrict__ dst,
                             int K, int N, int total) {
  const int t = blockIdx.x * blockDim.x + threadIdx.x;
  if (t >= total) return;
  const int kn  = K * N;
  const int mat = t / kn;
  const int rem = t - mat * kn;
  const int k = rem / N;
  const int n = rem - k * N;
  dst[(size_t)mat * kn + (size_t)n * K + k] = (_Float16)src[t];
}

// ---------------------------------------------------------------------------
// GEMM: C[M,N] = act(A_f32[M,K] @ W + bias[N]), W given transposed (N x K f16)
// 64x64 block tile, BK=32, 4 waves; wave w -> rows [w*16, w*16+16), all 64 cols.
// ---------------------------------------------------------------------------
template <int ACT, bool OUTF16>
__global__ void gemm_tile_kernel(const float* __restrict__ A,
                                 const _Float16* __restrict__ Bt,  // N x K
                                 const float* __restrict__ bias,
                                 float* __restrict__ Cf,
                                 _Float16* __restrict__ Ch,
                                 int M, int N, int K) {
  __shared__ _Float16 smem[4096];
  _Float16* Bsh = smem;          // [n][k] 64x32, LDS offset 0 (TDM target)
  _Float16* Ash = smem + 2048;   // [m][k] 64x32

  const int tid  = threadIdx.x;
  const int lane = tid & 31;
  const int wave = tid >> 5;
  const int row0 = blockIdx.y * 64;
  const int col0 = blockIdx.x * 64;

  const int l15  = lane & 15;
  const int hi   = lane >> 4;
  const int asel = hi << 4;      // A frag: hi lanes +16 in K
  const int bsel = hi << 4;      // B frag: hi lanes +16 in K

  v8f acc[4] = {{}, {}, {}, {}};

  for (int k0 = 0; k0 < K; k0 += 32) {
    // ---------------- stage A tile (64x32 f32 -> f16) ----------------
    {
      const int ar = tid >> 3;           // 0..15
      const int ak = (tid & 7) * 4;      // 0,4,...,28
#pragma unroll
      for (int j = 0; j < 4; ++j) {
        const int r = row0 + ar + j * 16;
        float x0 = 0.f, x1 = 0.f, x2 = 0.f, x3 = 0.f;
        if (r < M) {
          const float* src = A + (size_t)r * K + k0 + ak;
          if (k0 + 32 < K) __builtin_prefetch(src + 32, 0, 0);
          if (k0 + ak + 4 <= K) {
            const v4f v = *(const v4f*)src;
            x0 = v[0]; x1 = v[1]; x2 = v[2]; x3 = v[3];
          } else {
            if (k0 + ak + 0 < K) x0 = src[0];
            if (k0 + ak + 1 < K) x1 = src[1];
            if (k0 + ak + 2 < K) x2 = src[2];
            if (k0 + ak + 3 < K) x3 = src[3];
          }
        }
        _Float16* d = Ash + (ar + j * 16) * 32 + ak;
        d[0] = (_Float16)x0; d[1] = (_Float16)x1;
        d[2] = (_Float16)x2; d[3] = (_Float16)x3;
      }
    }

    // ---------------- stage B tile (64 cols x 32 k, f16) ----------------
#if HAVE_TDM
    if (wave == 0) {
      // Tensor DMA descriptor: 2-D tile of the N x K (f16) weight matrix.
      // tile: dim0 = 32 (K, contiguous), dim1 = 64 (N rows); OOB -> zero fill.
      const unsigned long long gab =
          (unsigned long long)(const void*)(Bt + (size_t)col0 * K + k0);
      const v4u g0 = {
          1u,                                            // count = 1
          0u,                                            // lds_addr (Bsh @ 0)
          (unsigned)(gab & 0xffffffffu),
          (unsigned)(((gab >> 32) & 0x01ffffffu) | (2u << 30))  // type = 2
      };
      const v8i g1 = {
          (int)(1u << 16),                               // data_size = 2 bytes
          (int)(((unsigned)K & 0xffffu) << 16),          // tensor_dim0 lo
          (int)((((unsigned)K >> 16) & 0xffffu) |
                (((unsigned)N & 0xffffu) << 16)),        // dim0 hi | dim1 lo
          (int)((((unsigned)N >> 16) & 0xffffu) |
                (32u << 16)),                            // dim1 hi | tile_dim0
          64,                                            // tile_dim1 (tile_dim2=0)
          (int)K,                                        // tensor_dim0_stride lo
          (int)(((unsigned)K & 0xffffu) << 16),          // stride0 hi | stride1 lo
          (int)(((unsigned)K >> 16) & 0xffffu)           // stride1 hi
      };
      const v4i gz = {0, 0, 0, 0};
#if defined(__clang_major__) && (__clang_major__ >= 23)
      const v8i gz8 = {0, 0, 0, 0, 0, 0, 0, 0};
      __builtin_amdgcn_tensor_load_to_lds(g0, g1, gz, gz, gz8, 0);
#else
      __builtin_amdgcn_tensor_load_to_lds(g0, g1, gz, gz, 0);
#endif
      __builtin_amdgcn_s_wait_tensorcnt(0);
    }
#else
    {
      const int bn = tid >> 2;           // 0..31
      const int bk = (tid & 3) * 8;      // 0,8,16,24
#pragma unroll
      for (int j = 0; j < 2; ++j) {
        const int n  = bn + j * 32;
        const int gn = col0 + n;
        _Float16* d = Bsh + n * 32 + bk;
        if (gn < N && k0 + bk + 8 <= K) {
          *(v8h*)d = *(const v8h*)(Bt + (size_t)gn * K + k0 + bk);
        } else {
#pragma unroll
          for (int e = 0; e < 8; ++e)
            d[e] = (gn < N && k0 + bk + e < K)
                     ? Bt[(size_t)gn * K + k0 + bk + e] : (_Float16)0.0f;
        }
      }
    }
#endif
    __syncthreads();

    // ---------------- fragments + 4 WMMA ----------------
    const int ar = wave * 16 + l15;
    const v8h alo = *(const v8h*)(Ash + ar * 32 + asel);
    const v8h ahi = *(const v8h*)(Ash + ar * 32 + asel + 8);
    const v16h afrag = __builtin_shufflevector(
        alo, ahi, 0, 1, 2, 3, 4, 5, 6, 7, 8, 9, 10, 11, 12, 13, 14, 15);
#pragma unroll
    for (int j = 0; j < 4; ++j) {
      const int cb = j * 16 + l15;
      const v16h bfrag = *(const v16h*)(Bsh + cb * 32 + bsel);
      acc[j] = __builtin_amdgcn_wmma_f32_16x16x32_f16(
          false, afrag, false, bfrag, (short)0, acc[j], false, false);
    }
    __syncthreads();
  }

  // ---------------- epilogue ----------------
  const int mbase = row0 + wave * 16 + (hi << 3);
  const bool full = (row0 + 64 <= M) && (col0 + 64 <= N);
#pragma unroll
  for (int j = 0; j < 4; ++j) {
    const int c = col0 + j * 16 + l15;
    const bool colok = (c < N);
    const float bv = (bias != nullptr && colok) ? bias[c] : 0.0f;
    if (full) {
#pragma unroll
      for (int rr = 0; rr < 8; ++rr) {
        float v = acc[j][rr] + bv;
        if (ACT == 1) v = fmaxf(v, 0.0f);
        const size_t o = (size_t)(mbase + rr) * N + c;
        if (OUTF16) Ch[o] = (_Float16)v;
        else        Cf[o] = v;
      }
    } else {
#pragma unroll
      for (int rr = 0; rr < 8; ++rr) {
        const int m = mbase + rr;
        if (m < M && colok) {
          float v = acc[j][rr] + bv;
          if (ACT == 1) v = fmaxf(v, 0.0f);
          if (OUTF16) Ch[(size_t)m * N + c] = (_Float16)v;
          else        Cf[(size_t)m * N + c] = v;
        }
      }
    }
  }
}

// ---------------------------------------------------------------------------
// Attention: S[b,h] = (Qh * scale) @ Kh^T   (K = 32, one WMMA per tile)
// qh/kh layout: [(b*300+q)*256 + h*32 + d]
// ---------------------------------------------------------------------------
__global__ void attn_qk_kernel(const float* __restrict__ qh,
                               const float* __restrict__ kh,
                               float* __restrict__ S, float scale) {
  const int lane = threadIdx.x & 31;
  const int wave = threadIdx.x >> 5;
  const int row0 = (blockIdx.y * 4 + wave) * 16;
  const int col0 = blockIdx.x * 16;
  const int bh = blockIdx.z;
  const int b = bh >> 3, h = bh & 7;
  if (row0 >= NQ) return;

  const int  r     = row0 + (lane & 15);
  const int  c     = col0 + (lane & 15);
  const bool rowok = (r < NQ);
  const bool colok = (c < NQ);
  const float* qrow = qh + ((size_t)(b * NQ + (rowok ? r : 0))) * DMODEL + h * HD;
  const float* krow = kh + ((size_t)(b * NQ + (colok ? c : 0))) * DMODEL + h * HD;
  const int asel = (lane >> 4) << 4;
  const int bsel = (lane >> 4) << 4;

  v16h a, bf;
#pragma unroll
  for (int i = 0; i < 8; ++i) {
    const int kp = 2 * i;
    const int k  = asel + kp;
    a[2 * i]     = (_Float16)(rowok ? qrow[k]     * scale : 0.0f);
    a[2 * i + 1] = (_Float16)(rowok ? qrow[k + 1] * scale : 0.0f);
  }
#pragma unroll
  for (int i = 0; i < 8; ++i) {
    const int k = bsel + 2 * i;
    bf[2 * i]     = (_Float16)(colok ? krow[k]     : 0.0f);
    bf[2 * i + 1] = (_Float16)(colok ? krow[k + 1] : 0.0f);
  }
  v8f acc = {};
  acc = __builtin_amdgcn_wmma_f32_16x16x32_f16(false, a, false, bf,
                                               (short)0, acc, false, false);
  float* Sp = S + (size_t)bh * NQ * NQ;
  const int mbase = row0 + ((lane >> 4) << 3);
#pragma unroll
  for (int rr = 0; rr < 8; ++rr) {
    const int m = mbase + rr;
    if (m < NQ && colok) Sp[(size_t)m * NQ + c] = acc[rr];
  }
}

// O[b,h] = S[b,h] (300x300) @ Vh (300x32); output [(b*300+q)*256 + h*32 + d]
__global__ void attn_sv_kernel(const float* __restrict__ S,
                               const float* __restrict__ vh,
                               float* __restrict__ O) {
  const int lane = threadIdx.x & 31;
  const int wave = threadIdx.x >> 5;
  const int row0 = (blockIdx.y * 4 + wave) * 16;
  const int col0 = blockIdx.x * 16;           // 0 or 16 (N = 32)
  const int bh = blockIdx.z;
  const int b = bh >> 3, h = bh & 7;
  if (row0 >= NQ) return;

  const int  r     = row0 + (lane & 15);
  const bool rowok = (r < NQ);
  const int  c     = col0 + (lane & 15);      // always < 32
  const float* Srow = S + ((size_t)bh * NQ + (rowok ? r : 0)) * NQ;
  const int asel = (lane >> 4) << 4;
  const int bsel = (lane >> 4) << 4;

  v8f acc = {};
  for (int k0 = 0; k0 < NQ; k0 += 32) {
    v16h a, bf;
#pragma unroll
    for (int i = 0; i < 8; ++i) {
      const int kp = 2 * i;
      const int k  = k0 + asel + kp;
      a[2 * i]     = (_Float16)((rowok && k     < NQ) ? Srow[k]     : 0.0f);
      a[2 * i + 1] = (_Float16)((rowok && k + 1 < NQ) ? Srow[k + 1] : 0.0f);
    }
#pragma unroll
    for (int i = 0; i < 8; ++i) {
      const int k = k0 + bsel + 2 * i;
      const float y0 = (k     < NQ) ? vh[((size_t)(b * NQ + k))     * DMODEL + h * HD + c] : 0.0f;
      const float y1 = (k + 1 < NQ) ? vh[((size_t)(b * NQ + k + 1)) * DMODEL + h * HD + c] : 0.0f;
      bf[2 * i]     = (_Float16)y0;
      bf[2 * i + 1] = (_Float16)y1;
    }
    acc = __builtin_amdgcn_wmma_f32_16x16x32_f16(false, a, false, bf,
                                                 (short)0, acc, false, false);
  }
  const int mbase = row0 + ((lane >> 4) << 3);
#pragma unroll
  for (int rr = 0; rr < 8; ++rr) {
    const int m = mbase + rr;
    if (m < NQ) O[((size_t)(b * NQ + m)) * DMODEL + h * HD + c] = acc[rr];
  }
}

// ---------------------------------------------------------------------------
// Row softmax over 300 (one wave per row)
// ---------------------------------------------------------------------------
__global__ void softmax300_kernel(float* __restrict__ S, int nrows) {
  const int lane = threadIdx.x & 31;
  const int wave = threadIdx.x >> 5;
  const int row = blockIdx.x * 8 + wave;
  if (row >= nrows) return;
  float* p = S + (size_t)row * NQ;
  float mx = -1e30f;
  for (int j = lane; j < NQ; j += 32) mx = fmaxf(mx, p[j]);
  for (int o = 16; o; o >>= 1) mx = fmaxf(mx, __shfl_xor(mx, o, 32));
  float sum = 0.0f;
  for (int j = lane; j < NQ; j += 32) {
    const float e = __expf(p[j] - mx);
    p[j] = e;
    sum += e;
  }
  for (int o = 16; o; o >>= 1) sum += __shfl_xor(sum, o, 32);
  const float inv = 1.0f / sum;
  for (int j = lane; j < NQ; j += 32) p[j] *= inv;
}

// softmax over 12 attention weights per (b,q,h) — one thread each
__global__ void softmax12_kernel(float* __restrict__ aw, int n) {
  const int t = blockIdx.x * blockDim.x + threadIdx.x;
  if (t >= n) return;
  const int row = t >> 3, h = t & 7;
  float* p = aw + (size_t)row * (NHEAD * NLVL * NPT) + h * (NLVL * NPT);
  float mx = -1e30f;
#pragma unroll
  for (int j = 0; j < 12; ++j) mx = fmaxf(mx, p[j]);
  float s = 0.0f;
#pragma unroll
  for (int j = 0; j < 12; ++j) { const float e = __expf(p[j] - mx); p[j] = e; s += e; }
  const float inv = 1.0f / s;
#pragma unroll
  for (int j = 0; j < 12; ++j) p[j] *= inv;
}

// ---------------------------------------------------------------------------
// y = LayerNorm(x + d) * g + b   (row = 256, one wave per row, in-place safe)
// ---------------------------------------------------------------------------
__global__ void add_ln_kernel(const float* __restrict__ x,
                              const float* __restrict__ d,
                              const float* __restrict__ g,
                              const float* __restrict__ b,
                              float* __restrict__ y, int nrows) {
  const int lane = threadIdx.x & 31;
  const int wave = threadIdx.x >> 5;
  const int row = blockIdx.x * 8 + wave;
  if (row >= nrows) return;
  const float* xr = x + (size_t)row * DMODEL;
  const float* dr = d + (size_t)row * DMODEL;
  float v[8];
  float s = 0.0f;
#pragma unroll
  for (int i = 0; i < 8; ++i) {
    v[i] = xr[lane + 32 * i] + dr[lane + 32 * i];
    s += v[i];
  }
  for (int o = 16; o; o >>= 1) s += __shfl_xor(s, o, 32);
  const float mean = s * (1.0f / DMODEL);
  float vs = 0.0f;
#pragma unroll
  for (int i = 0; i < 8; ++i) { const float t = v[i] - mean; vs += t * t; }
  for (int o = 16; o; o >>= 1) vs += __shfl_xor(vs, o, 32);
  const float rstd = rsqrtf(vs * (1.0f / DMODEL) + 1e-5f);
  float* yr = y + (size_t)row * DMODEL;
#pragma unroll
  for (int i = 0; i < 8; ++i) {
    const int col = lane + 32 * i;
    yr[col] = (v[i] - mean) * rstd * g[col] + b[col];
  }
}

// ---------------------------------------------------------------------------
// elementwise helpers
// ---------------------------------------------------------------------------
__global__ void vadd_kernel(const float* __restrict__ a,
                            const float* __restrict__ b,
                            float* __restrict__ c, int n) {
  const int t = blockIdx.x * blockDim.x + threadIdx.x;
  if (t < n) c[t] = a[t] + b[t];
}

__global__ void sigmoid_kernel(const float* __restrict__ a,
                               float* __restrict__ c, int n) {
  const int t = blockIdx.x * blockDim.x + threadIdx.x;
  if (t < n) c[t] = 1.0f / (1.0f + __expf(-a[t]));
}

// qpos hidden: h1 = relu(ref_d(4800x4) @ Wp1(4x512) + bp1)
__global__ void qpos_h1_kernel(const float* __restrict__ refd,
                               const float* __restrict__ Wp1,
                               const float* __restrict__ bp1,
                               float* __restrict__ h1) {
  const int t = blockIdx.x * blockDim.x + threadIdx.x;
  if (t >= MROWS * 2 * DMODEL) return;
  const int m = t >> 9, n = t & 511;
  const float* r = refd + (size_t)m * 4;
  float acc = bp1[n];
#pragma unroll
  for (int k = 0; k < 4; ++k) acc += r[k] * Wp1[k * (2 * DMODEL) + n];
  h1[t] = fmaxf(acc, 0.0f);
}

// ref update: new = sigmoid(delta + inv_sigmoid(clip(ref)))
__global__ void update_ref_kernel(const float* __restrict__ delta,
                                  const float* __restrict__ refd,
                                  float* __restrict__ refnew,
                                  float* __restrict__ bbox_out,
                                  int write_out, int n) {
  const int t = blockIdx.x * blockDim.x + threadIdx.x;
  if (t >= n) return;
  float r = fminf(fmaxf(refd[t], 0.0f), 1.0f);
  const float inv = __logf(fmaxf(r, 1e-5f)) - __logf(fmaxf(1.0f - r, 1e-5f));
  const float s = 1.0f / (1.0f + __expf(-(delta[t] + inv)));
  refnew[t] = s;
  if (write_out) bbox_out[t] = s;
}

// ---------------------------------------------------------------------------
// Deformable attention sampling: one wave per (b,q,h); lane = channel d.
// val (f16): [(b*8400 + loc)*256 + h*32 + d]
// ---------------------------------------------------------------------------
__global__ void deform_sample_kernel(const _Float16* __restrict__ val,
                                     const float* __restrict__ refd,
                                     const float* __restrict__ off,
                                     const float* __restrict__ aw,
                                     float* __restrict__ dout) {
  const int lane = threadIdx.x & 31;
  const int wave = threadIdx.x >> 5;
  const int idx = blockIdx.x * 8 + wave;      // (b*300+q)*8 + h
  if (idx >= MROWS * NHEAD) return;
  const int h = idx & 7;
  const int bq = idx >> 3;
  const int b = bq / NQ;

  const float* rp = refd + (size_t)bq * 4;
  const float rx = rp[0], ry = rp[1], rw = rp[2], rh = rp[3];
  const float* offp = off + (size_t)bq * (NHEAD * NLVL * NPT * 2) + h * (NLVL * NPT * 2);
  const float* awp  = aw  + (size_t)bq * (NHEAD * NLVL * NPT)     + h * (NLVL * NPT);

  const int DIMS[NLVL]   = {80, 40, 20};
  const int STARTS[NLVL] = {0, 6400, 8000};

  float acc = 0.0f;
#pragma unroll
  for (int lvl = 0; lvl < NLVL; ++lvl) {
    const int Wd = DIMS[lvl], Hd = DIMS[lvl], s0 = STARTS[lvl];
    const _Float16* vb = val + ((size_t)b * LVTOT + s0) * DMODEL + h * HD + lane;
#pragma unroll
    for (int p = 0; p < NPT; ++p) {
      const float ox = offp[(lvl * NPT + p) * 2 + 0];
      const float oy = offp[(lvl * NPT + p) * 2 + 1];
      const float lx = rx + ox * 0.125f * rw;     // ox/P * rw * 0.5
      const float ly = ry + oy * 0.125f * rh;
      const float x = lx * (float)Wd - 0.5f;
      const float y = ly * (float)Hd - 0.5f;
      const float fx = floorf(x), fy = floorf(y);
      const int x0 = (int)fx, y0 = (int)fy;
      const float wx = x - fx, wy = y - fy;
      const float a = awp[lvl * NPT + p];
      float sample = 0.0f;
#pragma unroll
      for (int dy = 0; dy < 2; ++dy) {
#pragma unroll
        for (int dx = 0; dx < 2; ++dx) {
          const int xi = x0 + dx, yi = y0 + dy;
          const float wgt = (dx ? wx : 1.0f - wx) * (dy ? wy : 1.0f - wy);
          if (xi >= 0 && xi < Wd && yi >= 0 && yi < Hd) {
            sample += wgt * (float)vb[((size_t)(yi * Wd + xi)) * DMODEL];
          }
        }
      }
      acc += a * sample;
    }
  }
  dout[(size_t)bq * DMODEL + h * HD + lane] = acc;
}

// ---------------------------------------------------------------------------
// host launcher
// ---------------------------------------------------------------------------
extern "C" void kernel_launch(void* const* d_in, const int* in_sizes, int n_in,
                              void* d_out, int out_size, void* d_ws, size_t ws_size,
                              hipStream_t stream) {
  const float* tgt   = (const float*)d_in[0];
  const float* refpu = (const float*)d_in[1];
  const float* mem   = (const float*)d_in[2];
  const float* Wq   = (const float*)d_in[5];   const float* bq   = (const float*)d_in[6];
  const float* Wk   = (const float*)d_in[7];   const float* bk   = (const float*)d_in[8];
  const float* Wv   = (const float*)d_in[9];   const float* bv   = (const float*)d_in[10];
  const float* Wo   = (const float*)d_in[11];  const float* bo   = (const float*)d_in[12];
  const float* ln1g = (const float*)d_in[13];  const float* ln1b = (const float*)d_in[14];
  const float* ln2g = (const float*)d_in[15];  const float* ln2b = (const float*)d_in[16];
  const float* ln3g = (const float*)d_in[17];  const float* ln3b = (const float*)d_in[18];
  const float* Wvp  = (const float*)d_in[19];  const float* bvp  = (const float*)d_in[20];
  const float* Woff = (const float*)d_in[21];  const float* boff = (const float*)d_in[22];
  const float* Waw  = (const float*)d_in[23];  const float* baw  = (const float*)d_in[24];
  const float* Wop  = (const float*)d_in[25];  const float* bop  = (const float*)d_in[26];
  const float* Wff1 = (const float*)d_in[27];  const float* bff1 = (const float*)d_in[28];
  const float* Wff2 = (const float*)d_in[29];  const float* bff2 = (const float*)d_in[30];
  const float* Wb1  = (const float*)d_in[31];  const float* bb1  = (const float*)d_in[32];
  const float* Wb2  = (const float*)d_in[33];  const float* bb2  = (const float*)d_in[34];
  const float* Wb3  = (const float*)d_in[35];  const float* bb3  = (const float*)d_in[36];
  const float* Ws   = (const float*)d_in[37];  const float* bs_  = (const float*)d_in[38];
  const float* Wp1  = (const float*)d_in[39];  const float* bp1  = (const float*)d_in[40];
  const float* Wp2  = (const float*)d_in[41];  const float* bp2  = (const float*)d_in[42];

  float* out_f = (float*)d_out;                // bboxes[19200] ++ logits[384000]
  float* bbox_out  = out_f;
  float* logit_out = out_f + BS * NQ * 4;

  size_t woff = 0;
  auto alloc = [&](size_t bytes) -> void* {
    woff = (woff + 255) & ~(size_t)255;
    void* p = (char*)d_ws + woff;
    woff += bytes;
    return p;
  };

  const int DD = DMODEL * DMODEL;              // 65536
  // transposed f16 weights (per-layer matrices contiguous, each N x K)
  _Float16* Wqh   = (_Float16*)alloc((size_t)NLAYER * DD * 2);
  _Float16* Wkh   = (_Float16*)alloc((size_t)NLAYER * DD * 2);
  _Float16* Wvh   = (_Float16*)alloc((size_t)NLAYER * DD * 2);
  _Float16* Woh   = (_Float16*)alloc((size_t)NLAYER * DD * 2);
  _Float16* Wvph  = (_Float16*)alloc((size_t)NLAYER * DD * 2);
  _Float16* Woph  = (_Float16*)alloc((size_t)NLAYER * DD * 2);
  _Float16* Wb1h  = (_Float16*)alloc((size_t)NLAYER * DD * 2);
  _Float16* Wb2h  = (_Float16*)alloc((size_t)NLAYER * DD * 2);
  _Float16* Wff1h = (_Float16*)alloc((size_t)NLAYER * DMODEL * DFF * 2);
  _Float16* Wff2h = (_Float16*)alloc((size_t)NLAYER * DFF * DMODEL * 2);
  _Float16* Woffh = (_Float16*)alloc((size_t)NLAYER * DMODEL * 192 * 2);
  _Float16* Wawh  = (_Float16*)alloc((size_t)NLAYER * DMODEL * 96 * 2);
  _Float16* Wb3h  = (_Float16*)alloc((size_t)NLAYER * DMODEL * 4 * 2);
  _Float16* Wsh   = (_Float16*)alloc((size_t)NLAYER * DMODEL * NCLS * 2);
  _Float16* Wp2h  = (_Float16*)alloc((size_t)2 * DMODEL * DMODEL * 2);

  const size_t ROWB = (size_t)MROWS * DMODEL;  // 1,228,800
  float* outb  = (float*)alloc(ROWB * 4);
  float* qpos  = (float*)alloc(ROWB * 4);
  float* h1    = (float*)alloc((size_t)MROWS * 2 * DMODEL * 4);
  float* qbuf  = (float*)alloc(ROWB * 4);
  float* qh    = (float*)alloc(ROWB * 4);
  float* kh    = (float*)alloc(ROWB * 4);
  float* vh    = (float*)alloc(ROWB * 4);
  float* Sb    = (float*)alloc((size_t)BS * NHEAD * NQ * NQ * 4);
  float* attO  = (float*)alloc(ROWB * 4);
  float* tmp1  = (float*)alloc(ROWB * 4);
  _Float16* valh = (_Float16*)alloc((size_t)BS * LVTOT * DMODEL * 2);
  float* offb  = (float*)alloc((size_t)MROWS * 192 * 4);
  float* awb   = (float*)alloc((size_t)MROWS * 96 * 4);
  float* doutb = (float*)alloc(ROWB * 4);
  float* ffh   = (float*)alloc((size_t)MROWS * DFF * 4);
  float* bbh1  = (float*)alloc(ROWB * 4);
  float* bbh2  = (float*)alloc(ROWB * 4);
  float* delta = (float*)alloc((size_t)MROWS * 4 * 4);
  float* refA  = (float*)alloc((size_t)MROWS * 4 * 4);
  float* refB  = (float*)alloc((size_t)MROWS * 4 * 4);

  auto cvt_t = [&](const float* src, _Float16* dst, int K, int N, int nmat) {
    const int total = nmat * K * N;
    cvt_t_kernel<<<(total + 255) / 256, 256, 0, stream>>>(src, dst, K, N, total);
  };
  auto gemm = [&](const float* A, const _Float16* B, const float* bias, float* C,
                  int M, int N, int K, int relu) {
    dim3 grid((N + 63) / 64, (M + 63) / 64);
    if (relu)
      gemm_tile_kernel<1, false><<<grid, 128, 0, stream>>>(A, B, bias, C, nullptr, M, N, K);
    else
      gemm_tile_kernel<0, false><<<grid, 128, 0, stream>>>(A, B, bias, C, nullptr, M, N, K);
  };
  auto gemm_h = [&](const float* A, const _Float16* B, const float* bias, _Float16* C,
                    int M, int N, int K) {
    dim3 grid((N + 63) / 64, (M + 63) / 64);
    gemm_tile_kernel<0, true><<<grid, 128, 0, stream>>>(A, B, bias, nullptr, C, M, N, K);
  };

  // one-time weight transposition/conversion + state init
  cvt_t(Wq,   Wqh,   DMODEL, DMODEL, NLAYER);
  cvt_t(Wk,   Wkh,   DMODEL, DMODEL, NLAYER);
  cvt_t(Wv,   Wvh,   DMODEL, DMODEL, NLAYER);
  cvt_t(Wo,   Woh,   DMODEL, DMODEL, NLAYER);
  cvt_t(Wvp,  Wvph,  DMODEL, DMODEL, NLAYER);
  cvt_t(Wop,  Woph,  DMODEL, DMODEL, NLAYER);
  cvt_t(Wb1,  Wb1h,  DMODEL, DMODEL, NLAYER);
  cvt_t(Wb2,  Wb2h,  DMODEL, DMODEL, NLAYER);
  cvt_t(Wff1, Wff1h, DMODEL, DFF,    NLAYER);
  cvt_t(Wff2, Wff2h, DFF,    DMODEL, NLAYER);
  cvt_t(Woff, Woffh, DMODEL, 192,    NLAYER);
  cvt_t(Waw,  Wawh,  DMODEL, 96,     NLAYER);
  cvt_t(Wb3,  Wb3h,  DMODEL, 4,      NLAYER);
  cvt_t(Ws,   Wsh,   DMODEL, NCLS,   NLAYER);
  cvt_t(Wp2,  Wp2h,  2 * DMODEL, DMODEL, 1);

  hipMemcpyAsync(outb, tgt, ROWB * 4, hipMemcpyDeviceToDevice, stream);
  sigmoid_kernel<<<(MROWS * 4 + 255) / 256, 256, 0, stream>>>(refpu, refA, MROWS * 4);

  float* refCur = refA;
  float* refNxt = refB;
  const float scale = 0.17677669529663687f;    // 1/sqrt(32)

  for (int i = 0; i < NLAYER; ++i) {
    // query pos embedding
    qpos_h1_kernel<<<(MROWS * 2 * DMODEL + 255) / 256, 256, 0, stream>>>(refCur, Wp1, bp1, h1);
    gemm(h1, Wp2h, bp2, qpos, MROWS, DMODEL, 2 * DMODEL, 0);
    vadd_kernel<<<(int)((ROWB + 255) / 256), 256, 0, stream>>>(outb, qpos, qbuf, (int)ROWB);

    // self-attention
    gemm(qbuf, Wqh + (size_t)i * DD, bq + i * DMODEL, qh, MROWS, DMODEL, DMODEL, 0);
    gemm(qbuf, Wkh + (size_t)i * DD, bk + i * DMODEL, kh, MROWS, DMODEL, DMODEL, 0);
    gemm(outb, Wvh + (size_t)i * DD, bv + i * DMODEL, vh, MROWS, DMODEL, DMODEL, 0);
    {
      dim3 g1((NQ + 15) / 16, (NQ + 63) / 64, BS * NHEAD);
      attn_qk_kernel<<<g1, 128, 0, stream>>>(qh, kh, Sb, scale);
      softmax300_kernel<<<(BS * NHEAD * NQ + 7) / 8, 256, 0, stream>>>(Sb, BS * NHEAD * NQ);
      dim3 g2(HD / 16, (NQ + 63) / 64, BS * NHEAD);
      attn_sv_kernel<<<g2, 128, 0, stream>>>(Sb, vh, attO);
    }
    gemm(attO, Woh + (size_t)i * DD, bo + i * DMODEL, tmp1, MROWS, DMODEL, DMODEL, 0);
    add_ln_kernel<<<(MROWS + 7) / 8, 256, 0, stream>>>(outb, tmp1,
        ln1g + i * DMODEL, ln1b + i * DMODEL, outb, MROWS);

    // deformable attention
    vadd_kernel<<<(int)((ROWB + 255) / 256), 256, 0, stream>>>(outb, qpos, qbuf, (int)ROWB);
    gemm_h(mem, Wvph + (size_t)i * DD, bvp + i * DMODEL, valh, BS * LVTOT, DMODEL, DMODEL);
    gemm(qbuf, Woffh + (size_t)i * DMODEL * 192, boff + i * 192, offb, MROWS, 192, DMODEL, 0);
    gemm(qbuf, Wawh + (size_t)i * DMODEL * 96,  baw  + i * 96,  awb,  MROWS, 96,  DMODEL, 0);
    softmax12_kernel<<<(MROWS * NHEAD + 255) / 256, 256, 0, stream>>>(awb, MROWS * NHEAD);
    deform_sample_kernel<<<(MROWS * NHEAD + 7) / 8, 256, 0, stream>>>(valh, refCur, offb, awb, doutb);
    gemm(doutb, Woph + (size_t)i * DD, bop + i * DMODEL, tmp1, MROWS, DMODEL, DMODEL, 0);
    add_ln_kernel<<<(MROWS + 7) / 8, 256, 0, stream>>>(outb, tmp1,
        ln2g + i * DMODEL, ln2b + i * DMODEL, outb, MROWS);

    // FFN
    gemm(outb, Wff1h + (size_t)i * DMODEL * DFF, bff1 + i * DFF, ffh, MROWS, DFF, DMODEL, 1);
    gemm(ffh, Wff2h + (size_t)i * DFF * DMODEL, bff2 + i * DMODEL, tmp1, MROWS, DMODEL, DFF, 0);
    add_ln_kernel<<<(MROWS + 7) / 8, 256, 0, stream>>>(outb, tmp1,
        ln3g + i * DMODEL, ln3b + i * DMODEL, outb, MROWS);

    // bbox refinement head
    gemm(outb, Wb1h + (size_t)i * DD, bb1 + i * DMODEL, bbh1, MROWS, DMODEL, DMODEL, 1);
    gemm(bbh1, Wb2h + (size_t)i * DD, bb2 + i * DMODEL, bbh2, MROWS, DMODEL, DMODEL, 1);
    gemm(bbh2, Wb3h + (size_t)i * DMODEL * 4, bb3 + i * 4, delta, MROWS, 4, DMODEL, 0);
    const int last = (i == NLAYER - 1) ? 1 : 0;
    update_ref_kernel<<<(MROWS * 4 + 255) / 256, 256, 0, stream>>>(
        delta, refCur, refNxt, bbox_out, last, MROWS * 4);

    if (last) {
      gemm(outb, Wsh + (size_t)i * DMODEL * NCLS, bs_ + i * NCLS, logit_out,
           MROWS, NCLS, DMODEL, 0);
    }
    float* t = refCur; refCur = refNxt; refNxt = t;
  }
}
